// VectorDiscoBlock_6107443495089
// MI455X (gfx1250) — matrix operand order repaired and checked by reading, not compiled, not run on hardware
//
#include <hip/hip_runtime.h>
#include <cstdint>

// ---------------- problem constants ----------------
#define K_B   2
#define K_NS  128
#define K_NV  64
#define K_H   180
#define K_W   360
#define K_HID 256
#define K_TW  64          // pixels per workgroup tile
#define K_NXT 6           // ceil(360/64)
#define K_PW  66          // patch width (64 + 2 halo)

// padded per-pixel channel strides (stride-in-dwords == 4*odd mod 64 -> conflict-free
// ds_load_b128 per 16-lane phase; f32 buffers use odd-dword strides)
#define PS_S  136         // scalar patch / sGelu / s2bf16 (68 dwords)
#define PS_H  264         // hidden h (132 dwords)
#define PS_V  72          // vector patches (36 dwords)
#define PR_RES 129        // f32 s-residual (dwords per pixel)
#define PR_G   65         // f32 gate (dwords per pixel)

// ---------------- workspace layout ----------------
// Weight A-fragments: 32 lanes * 8 dwords = 256 dwords each, lane-contiguous.
#define NFRAG_SCONV (9*8*4)                       // 288
#define NFRAG_VCONV (9*4*2)                       // 72
#define OFF_SCONV 0u
#define OFF_WVS   (OFF_SCONV + NFRAG_SCONV*256u)
#define OFF_WVR   (OFF_WVS   + NFRAG_VCONV*256u)
#define OFF_WVRN  (OFF_WVR   + NFRAG_VCONV*256u)
#define OFF_MLP1  (OFF_WVRN  + NFRAG_VCONV*256u)
#define OFF_MLP2  (OFF_MLP1  + 64u*256u)
#define OFF_SV    (OFF_MLP2  + 64u*256u)
#define WS_WEIGHT_BYTES ((OFF_SV + 16u*256u) * 4u)          // 663552
// bf16 pixel-major activations (written by transpose prep kernel)
#define WS_XST_BYTES  ((size_t)K_B*K_H*K_W*K_NS*2)          // 33,177,600
#define WS_UVT_BYTES  ((size_t)K_B*K_H*K_W*K_NV*2)          // 16,588,800 each

// ---------------- LDS layout (bytes) ----------------
// Region A (reused): patchS bf16 [3][66][PS_S]            = 53856
//                    sGelu [64][PS_S]@0 (17408) + h [64][PS_H]@17408 (33792)
//                    patchU [3][66][PS_V]@0 (28512) + patchV @28512 (28512)
#define L_A     0
#define L_H     17408
#define L_PV    28512
#define L_SRES  57024     // f32 [64 px][PR_RES]  (33024)
#define L_S2BF  90048     // bf16 [64][PS_S]      (17408)
#define L_GATE  107456    // f32 [64 px][PR_G]    (16640)
#define L_MU    124096
#define L_RSTD  124352
#define L_TOTAL 124608

typedef __bf16 bf16x16 __attribute__((ext_vector_type(16)));
typedef float  f32x8   __attribute__((ext_vector_type(8)));

__device__ __forceinline__ unsigned short f2bf(float x) {
  unsigned int u = __float_as_uint(x);
  return (unsigned short)((u + 0x7FFFu + ((u >> 16) & 1u)) >> 16);
}

__device__ __forceinline__ float gelu_tanh(float x) {
  float x3 = x * x * x;
  return 0.5f * x * (1.0f + tanhf(0.7978845608028654f * (x + 0.044715f * x3)));
}

__device__ __forceinline__ f32x8 wmma_bf16(bf16x16 a, bf16x16 b, f32x8 c) {
  return __builtin_amdgcn_wmma_f32_16x16x32_bf16(false, a, false, b, (short)0, c,
                                                 false, false);
}

__device__ __forceinline__ bf16x16 ldA(const uint32_t* buf, int frag, int lane) {
  union { bf16x16 v; float4 f[2]; } u;
  const float4* p = (const float4*)(buf + (size_t)frag * 256 + lane * 8);
  u.f[0] = p[0];
  u.f[1] = p[1];
  return u.v;
}

__device__ __forceinline__ bf16x16 ldB(const char* smem, int byteOff) {
  union { bf16x16 v; float4 f[2]; } u;
  u.f[0] = *(const float4*)(smem + byteOff);
  u.f[1] = *(const float4*)(smem + byteOff + 16);
  return u.v;
}

// CDNA5 async copy: 16B global -> LDS, tracked by ASYNCcnt (no VGPR data path)
__device__ __forceinline__ void async_cp16(uint32_t lds_off, const void* gaddr) {
  asm volatile("global_load_async_to_lds_b128 %0, %1, off"
               :: "v"(lds_off), "v"((unsigned long long)gaddr)
               : "memory");
}
__device__ __forceinline__ void wait_async0() {
  asm volatile("s_wait_asynccnt 0x0" ::: "memory");
}

// ================= prep: activation transpose to pixel-major bf16 =================
__global__ void transpose_acts(const float* __restrict__ xs,
                               const float* __restrict__ xv,
                               unsigned short* __restrict__ xsT,
                               unsigned short* __restrict__ uT,
                               unsigned short* __restrict__ vT) {
  const size_t HW = (size_t)K_H * K_W;
  const size_t stride = (size_t)gridDim.x * blockDim.x;
  size_t totS = (size_t)K_B * HW * K_NS;
  for (size_t i = (size_t)blockIdx.x * blockDim.x + threadIdx.x; i < totS; i += stride) {
    int c = (int)(i & (K_NS - 1));
    size_t pix = i >> 7;                 // (b*H + y)*W + x
    int b = (int)(pix / HW);
    size_t rem = pix % HW;
    xsT[i] = f2bf(xs[(size_t)(b * K_NS + c) * HW + rem]);
  }
  size_t totV = (size_t)K_B * HW * K_NV;
  for (size_t i = (size_t)blockIdx.x * blockDim.x + threadIdx.x; i < totV; i += stride) {
    int c = (int)(i & (K_NV - 1));
    size_t pix = i >> 6;
    int b = (int)(pix / HW);
    size_t rem = pix % HW;
    const float* s = xv + ((size_t)(b * K_NV + c) * HW + rem) * 2;
    uT[i] = f2bf(s[0]);
    vT[i] = f2bf(s[1]);
  }
}

// ================= weight prep kernels =================
__global__ void pack_sconv(const float* __restrict__ cws,
                           const float* __restrict__ pwss,
                           uint32_t* __restrict__ dst) {
  int gid = blockIdx.x * blockDim.x + threadIdx.x;
  if (gid >= NFRAG_SCONV * 32) return;
  int f = gid >> 5, lane = gid & 31;
  int tap = f / 32, rem = f % 32, mt = rem >> 2, kb = rem & 3;
  int ky = tap / 3, kx = tap % 3;
  int m = lane & 15, kadd = (lane >> 4) ? 8 : 0;
  int o = mt * 16 + m;
#pragma unroll
  for (int d = 0; d < 8; ++d) {
    int k0 = 2 * kadd + d * 2;
    uint32_t pk = 0;
#pragma unroll
    for (int h = 0; h < 2; ++h) {
      int c = kb * 32 + k0 + h;
      float w = cws[((o * K_NS + c) * 3 + ky) * 3 + kx];
      if (ky == 1 && kx == 1) w += pwss[o * K_NS + c];
      pk |= ((uint32_t)f2bf(w)) << (16 * h);
    }
    dst[(size_t)f * 256 + lane * 8 + d] = pk;
  }
}

__global__ void pack_vconv(const float* __restrict__ cwv,
                           const float* __restrict__ pvt,
                           uint32_t* __restrict__ dWs, uint32_t* __restrict__ dWr,
                           uint32_t* __restrict__ dWrn) {
  int gid = blockIdx.x * blockDim.x + threadIdx.x;
  if (gid >= NFRAG_VCONV * 32) return;
  int f = gid >> 5, lane = gid & 31;
  int tap = f / 8, rem = f % 8, mt = rem >> 1, kb = rem & 1;
  int ky = tap / 3, kx = tap % 3;
  int m = lane & 15, kadd = (lane >> 4) ? 8 : 0;
  int o = mt * 16 + m;
#pragma unroll
  for (int d = 0; d < 8; ++d) {
    int k0 = 2 * kadd + d * 2;
    uint32_t ps = 0, pr = 0, pn = 0;
#pragma unroll
    for (int h = 0; h < 2; ++h) {
      int c = kb * 32 + k0 + h;
      float ws = cwv[(((o * K_NV + c) * 2 + 0) * 3 + ky) * 3 + kx];
      float wr = cwv[(((o * K_NV + c) * 2 + 1) * 3 + ky) * 3 + kx];
      if (ky == 1 && kx == 1) {
        ws += pvt[(o * K_NV + c) * 2 + 0];
        wr += pvt[(o * K_NV + c) * 2 + 1];
      }
      ps |= ((uint32_t)f2bf(ws)) << (16 * h);
      pr |= ((uint32_t)f2bf(wr)) << (16 * h);
      pn |= ((uint32_t)f2bf(-wr)) << (16 * h);
    }
    size_t idx = (size_t)f * 256 + lane * 8 + d;
    dWs[idx] = ps; dWr[idx] = pr; dWrn[idx] = pn;
  }
}

__global__ void pack_gemm(const float* __restrict__ src, uint32_t* __restrict__ dst,
                          int M, int Kc) {
  int nf = (M / 16) * (Kc / 32);
  int gid = blockIdx.x * blockDim.x + threadIdx.x;
  if (gid >= nf * 32) return;
  int f = gid >> 5, lane = gid & 31;
  int kt = Kc / 32, mt = f / kt, kb = f % kt;
  int m = lane & 15, kadd = (lane >> 4) ? 8 : 0;
  int o = mt * 16 + m;
#pragma unroll
  for (int d = 0; d < 8; ++d) {
    int k0 = 2 * kadd + d * 2;
    uint32_t pk = 0;
#pragma unroll
    for (int h = 0; h < 2; ++h) {
      int c = kb * 32 + k0 + h;
      pk |= ((uint32_t)f2bf(src[o * Kc + c])) << (16 * h);
    }
    dst[(size_t)f * 256 + lane * 8 + d] = pk;
  }
}

// ================= fused main kernel =================
__global__ __launch_bounds__(256, 1)
void disco_main(const float* __restrict__ xs, const float* __restrict__ xv,
                const unsigned short* __restrict__ xsT,
                const unsigned short* __restrict__ uT,
                const unsigned short* __restrict__ vT,
                const float* __restrict__ cbs, const float* __restrict__ b1,
                const float* __restrict__ b2, const float* __restrict__ svb,
                const float* __restrict__ lng, const float* __restrict__ lnb,
                const uint32_t* __restrict__ wsb,
                float* __restrict__ out_s, float* __restrict__ out_v) {
  extern __shared__ char smem[];
  const uint32_t ldsbase = (uint32_t)(uintptr_t)smem;   // low 32 bits = LDS offset
  const int tid = threadIdx.x;
  const int wave = tid >> 5, lane = tid & 31;
  const int laneN = lane & 15, laneHi = lane >> 4;
  const int chAdd = laneHi ? 8 : 0;

  const int xt = blockIdx.x % K_NXT;
  const int y  = (blockIdx.x / K_NXT) % K_H;
  const int b  = blockIdx.x / (K_NXT * K_H);
  const int x0 = xt * K_TW;

  // ---- phase 0: async-copy scalar patch (bf16, pixel-major, zero-fill halo rows) ----
  {
    const float4 z4 = {0.f, 0.f, 0.f, 0.f};
    for (int q = tid; q < 3 * K_PW * 16; q += 256) {     // 16B chunks
      int pix = q >> 4, w16 = q & 15;
      int r = pix / K_PW, p = pix % K_PW;
      int sy = y + r - 1;
      int sx = (x0 + p - 1 + K_W) % K_W;
      uint32_t loff = ldsbase + L_A + (uint32_t)(pix * (PS_S * 2) + w16 * 16);
      if (sy >= 0 && sy < K_H) {
        const char* src = (const char*)(xsT + (((size_t)(b * K_H + sy) * K_W + sx) << 7));
        async_cp16(loff, src + w16 * 16);
      } else {
        *(float4*)(smem + L_A + pix * (PS_S * 2) + w16 * 16) = z4;
      }
    }
  }
  wait_async0();
  __syncthreads();

  // ---- phase 1: scalar conv (9-tap implicit GEMM, pointwise folded into center) ----
  const int mt = wave;
  f32x8 accS[4];
#pragma unroll
  for (int i = 0; i < 4; ++i) accS[i] = (f32x8){0,0,0,0,0,0,0,0};

  for (int tap = 0; tap < 9; ++tap) {
    int ky = tap / 3, kx = tap % 3;
    if (tap < 8)   // prefetch next tap's A-fragments into cache
      __builtin_prefetch(wsb + OFF_SCONV + ((size_t)(tap + 1) * 32 + mt * 4) * 256, 0, 1);
#pragma unroll
    for (int kb = 0; kb < 4; ++kb) {
      bf16x16 a = ldA(wsb + OFF_SCONV, tap * 32 + mt * 4 + kb, lane);
#pragma unroll
      for (int nt = 0; nt < 4; ++nt) {
        int p = nt * 16 + laneN + kx;
        int off = L_A + ((ky * K_PW + p) * PS_S + kb * 32 + laneHi * 16) * 2;
        accS[nt] = wmma_bf16(a, ldB(smem, off), accS[nt]);
      }
    }
  }
  __syncthreads();

  // ---- phase 1b: +bias, GELU, stage sGelu bf16 [px][PS_S] ----
#pragma unroll
  for (int nt = 0; nt < 4; ++nt) {
    int px = nt * 16 + laneN;
#pragma unroll
    for (int r = 0; r < 8; ++r) {
      int ch = mt * 16 + r + chAdd;
      float v = gelu_tanh(accS[nt][r] + cbs[ch]);
      accS[nt][r] = v;
      *(unsigned short*)(smem + L_A + (px * PS_S + ch) * 2) = f2bf(v);
    }
  }
  __syncthreads();

  // ---- phase 2: MLP GEMM1 (256x128) -> h ----
  f32x8 accH[2][4];
#pragma unroll
  for (int j = 0; j < 2; ++j)
#pragma unroll
    for (int nt = 0; nt < 4; ++nt) accH[j][nt] = (f32x8){0,0,0,0,0,0,0,0};

#pragma unroll
  for (int j = 0; j < 2; ++j) {
    int mt1 = wave * 2 + j;
#pragma unroll
    for (int kb = 0; kb < 4; ++kb) {
      bf16x16 a = ldA(wsb + OFF_MLP1, mt1 * 4 + kb, lane);
#pragma unroll
      for (int nt = 0; nt < 4; ++nt) {
        int px = nt * 16 + laneN;
        int off = L_A + (px * PS_S + kb * 32 + laneHi * 16) * 2;
        accH[j][nt] = wmma_bf16(a, ldB(smem, off), accH[j][nt]);
      }
    }
  }
#pragma unroll
  for (int j = 0; j < 2; ++j) {
    int mt1 = wave * 2 + j;
#pragma unroll
    for (int nt = 0; nt < 4; ++nt) {
      int px = nt * 16 + laneN;
#pragma unroll
      for (int r = 0; r < 8; ++r) {
        int ch = mt1 * 16 + r + chAdd;
        float v = gelu_tanh(accH[j][nt][r] + b1[ch]);
        *(unsigned short*)(smem + L_H + (px * PS_H + ch) * 2) = f2bf(v);
      }
    }
  }
  __syncthreads();

  // ---- phase 3: MLP GEMM2 (128x256), C initialized with gelu(s) ----
#pragma unroll
  for (int kb = 0; kb < 8; ++kb) {
    bf16x16 a = ldA(wsb + OFF_MLP2, mt * 8 + kb, lane);
#pragma unroll
    for (int nt = 0; nt < 4; ++nt) {
      int px = nt * 16 + laneN;
      int off = L_H + (px * PS_H + kb * 32 + laneHi * 16) * 2;
      accS[nt] = wmma_bf16(a, ldB(smem, off), accS[nt]);
    }
  }
#pragma unroll
  for (int nt = 0; nt < 4; ++nt) {
    int px = nt * 16 + laneN;
    int gx = (x0 + px) % K_W;
#pragma unroll
    for (int r = 0; r < 8; ++r) {
      int ch = mt * 16 + r + chAdd;
      float s2 = accS[nt][r] + b2[ch];
      *(unsigned short*)(smem + L_S2BF + (px * PS_S + ch) * 2) = f2bf(s2);
      float res = s2 + xs[((size_t)(b * K_NS + ch) * K_H + y) * K_W + gx];
      *(float*)(smem + L_SRES + (px * PR_RES + ch) * 4) = res;
    }
  }
  __syncthreads();

  // ---- phase 4: gate GEMM + async vector patch + LN stats ----
  f32x8 accG[2];
#pragma unroll
  for (int j = 0; j < 2; ++j) accG[j] = (f32x8){0,0,0,0,0,0,0,0};
#pragma unroll
  for (int j = 0; j < 2; ++j) {
    int t = wave * 2 + j, mtg = t >> 2, nt = t & 3;
    int px = nt * 16 + laneN;
#pragma unroll
    for (int kb = 0; kb < 4; ++kb) {
      bf16x16 a = ldA(wsb + OFF_SV, mtg * 4 + kb, lane);
      int off = L_S2BF + (px * PS_S + kb * 32 + laneHi * 16) * 2;
      accG[j] = wmma_bf16(a, ldB(smem, off), accG[j]);
    }
  }
#pragma unroll
  for (int j = 0; j < 2; ++j) {
    int t = wave * 2 + j, mtg = t >> 2, nt = t & 3;
    int px = nt * 16 + laneN;
#pragma unroll
    for (int r = 0; r < 8; ++r) {
      int ch = mtg * 16 + r + chAdd;
      *(float*)(smem + L_GATE + (px * PR_G + ch) * 4) = accG[j][r] + svb[ch];
    }
  }
  // async-copy u/v patches into region A (h already consumed)
  {
    const float4 z4 = {0.f, 0.f, 0.f, 0.f};
    for (int q = tid; q < 3 * K_PW * 8; q += 256) {      // 16B chunks (128B per pixel)
      int pix = q >> 3, w16 = q & 7;
      int r = pix / K_PW, p = pix % K_PW;
      int sy = y + r - 1;
      int sx = (x0 + p - 1 + K_W) % K_W;
      uint32_t lu = ldsbase + L_A  + (uint32_t)(pix * (PS_V * 2) + w16 * 16);
      uint32_t lv = ldsbase + L_PV + (uint32_t)(pix * (PS_V * 2) + w16 * 16);
      if (sy >= 0 && sy < K_H) {
        size_t base = ((size_t)(b * K_H + sy) * K_W + sx) << 6;
        async_cp16(lu, (const char*)(uT + base) + w16 * 16);
        async_cp16(lv, (const char*)(vT + base) + w16 * 16);
      } else {
        *(float4*)(smem + L_A  + pix * (PS_V * 2) + w16 * 16) = z4;
        *(float4*)(smem + L_PV + pix * (PS_V * 2) + w16 * 16) = z4;
      }
    }
  }
  // LN stats per pixel (population variance)
  if (tid < K_TW) {
    const float* sr = (const float*)(smem + L_SRES) + tid * PR_RES;
    float sum = 0.f, sq = 0.f;
    for (int c = 0; c < K_NS; ++c) { float v = sr[c]; sum += v; sq += v * v; }
    float mu = sum * (1.0f / K_NS);
    float var = sq * (1.0f / K_NS) - mu * mu;
    ((float*)(smem + L_MU))[tid] = mu;
    ((float*)(smem + L_RSTD))[tid] = rsqrtf(var + 1e-6f);
  }
  wait_async0();
  __syncthreads();

  // ---- phase 5: store normalized scalar output ----
  for (int i = tid; i < K_NS * K_TW; i += 256) {
    int px = i & 63, ch = i >> 6;
    int gx = x0 + px;
    if (gx < K_W) {
      float v = ((const float*)(smem + L_SRES))[px * PR_RES + ch];
      float o = (v - ((const float*)(smem + L_MU))[px]) *
                ((const float*)(smem + L_RSTD))[px] * lng[ch] + lnb[ch];
      out_s[((size_t)(b * K_NS + ch) * K_H + y) * K_W + gx] = o;
    }
  }

  // ---- phase 6: complex vector conv; vu = ws*u + (-wr)*v ; vv = ws*v + wr*u ----
  f32x8 aU[2], aV[2];
#pragma unroll
  for (int j = 0; j < 2; ++j) { aU[j] = (f32x8){0,0,0,0,0,0,0,0}; aV[j] = aU[j]; }

  for (int tap = 0; tap < 9; ++tap) {
    int ky = tap / 3, kx = tap % 3;
    if (tap < 8)
      __builtin_prefetch(wsb + OFF_WVS + ((size_t)(tap + 1) * 8) * 256, 0, 1);
#pragma unroll
    for (int kb = 0; kb < 2; ++kb) {
#pragma unroll
      for (int j = 0; j < 2; ++j) {
        int t = wave * 2 + j, mtv = t >> 2, nt = t & 3;
        int fidx = tap * 8 + mtv * 2 + kb;
        bf16x16 aws = ldA(wsb + OFF_WVS, fidx, lane);
        bf16x16 awr = ldA(wsb + OFF_WVR, fidx, lane);
        bf16x16 awn = ldA(wsb + OFF_WVRN, fidx, lane);
        int p = nt * 16 + laneN + kx;
        int offU = L_A  + ((ky * K_PW + p) * PS_V + kb * 32 + laneHi * 16) * 2;
        int offV = L_PV + ((ky * K_PW + p) * PS_V + kb * 32 + laneHi * 16) * 2;
        bf16x16 bu = ldB(smem, offU);
        bf16x16 bv = ldB(smem, offV);
        aU[j] = wmma_bf16(aws, bu, aU[j]);
        aU[j] = wmma_bf16(awn, bv, aU[j]);
        aV[j] = wmma_bf16(aws, bv, aV[j]);
        aV[j] = wmma_bf16(awr, bu, aV[j]);
      }
    }
  }

  // ---- phase 7: vector output: x_vector + (conv + gate*field) ----
#pragma unroll
  for (int j = 0; j < 2; ++j) {
    int t = wave * 2 + j, mtv = t >> 2, nt = t & 3;
    int px = nt * 16 + laneN;
    int gx = x0 + px;
    if (gx < K_W) {
#pragma unroll
      for (int r = 0; r < 8; ++r) {
        int ch = mtv * 16 + r + chAdd;
        size_t base = (((size_t)(b * K_NV + ch) * K_H + y) * K_W + gx) * 2;
        float uu = xv[base], vv = xv[base + 1];
        float g = ((const float*)(smem + L_GATE))[px * PR_G + ch];
        out_v[base]     = uu + aU[j][r] + g * uu;
        out_v[base + 1] = vv + aV[j][r] + g * vv;
      }
    }
  }
}

// ================= launcher =================
extern "C" void kernel_launch(void* const* d_in, const int* in_sizes, int n_in,
                              void* d_out, int out_size, void* d_ws, size_t ws_size,
                              hipStream_t stream) {
  const float* xs   = (const float*)d_in[0];
  const float* xv   = (const float*)d_in[1];
  const float* cws  = (const float*)d_in[2];
  const float* cbs  = (const float*)d_in[3];
  const float* cwv  = (const float*)d_in[4];
  const float* pwss = (const float*)d_in[5];
  const float* pvt  = (const float*)d_in[6];
  const float* w1   = (const float*)d_in[7];
  const float* b1   = (const float*)d_in[8];
  const float* w2   = (const float*)d_in[9];
  const float* b2   = (const float*)d_in[10];
  const float* svw  = (const float*)d_in[11];
  const float* svb  = (const float*)d_in[12];
  const float* lng  = (const float*)d_in[13];
  const float* lnb  = (const float*)d_in[14];

  uint32_t* wsb = (uint32_t*)d_ws;
  unsigned short* xsT = (unsigned short*)((char*)d_ws + WS_WEIGHT_BYTES);
  unsigned short* uT  = (unsigned short*)((char*)xsT + WS_XST_BYTES);
  unsigned short* vT  = (unsigned short*)((char*)uT + WS_UVT_BYTES);

  float* out_s = (float*)d_out;
  float* out_v = out_s + (size_t)K_B * K_NS * K_H * K_W;

  pack_sconv<<<dim3((NFRAG_SCONV * 32 + 255) / 256), dim3(256), 0, stream>>>(
      cws, pwss, wsb + OFF_SCONV);
  pack_vconv<<<dim3((NFRAG_VCONV * 32 + 255) / 256), dim3(256), 0, stream>>>(
      cwv, pvt, wsb + OFF_WVS, wsb + OFF_WVR, wsb + OFF_WVRN);
  pack_gemm<<<dim3(8), dim3(256), 0, stream>>>(w1, wsb + OFF_MLP1, K_HID, K_NS);
  pack_gemm<<<dim3(8), dim3(256), 0, stream>>>(w2, wsb + OFF_MLP2, K_NS, K_HID);
  pack_gemm<<<dim3(2), dim3(256), 0, stream>>>(svw, wsb + OFF_SV, K_NV, K_NS);
  transpose_acts<<<dim3(4096), dim3(256), 0, stream>>>(xs, xv, xsT, uT, vT);

  dim3 grid(K_B * K_H * K_NXT);   // 2160 workgroups
  disco_main<<<grid, dim3(256), L_TOTAL, stream>>>(xs, xv, xsT, uT, vT,
                                                   cbs, b1, b2, svb, lng, lnb,
                                                   wsb, out_s, out_v);
}
